// GeometricReasoningOriginalImpl_10187662426413
// MI455X (gfx1250) — matrix operand order repaired and checked by reading, not compiled, not run on hardware
//
#include <hip/hip_runtime.h>

// Problem constants (B,S,C,H,M) = (2,384,1536,128,1)
#define Bsz   2
#define Ssz   384
#define Csz   1536
#define Hsz   128
#define DPROJ 1920           // 4*H*3 + H*3*M
#define NROT  1152           // H*3*(2+M)
#define AODIM 384            // H*3*M
#define ROWS  (Bsz*Ssz)      // 768
#define NEGF  (-3.4028234663852886e+38f)
#define INV_SQRT3 0.5773502691896258f
#define LN_EPS 1e-5f

typedef __attribute__((ext_vector_type(16))) _Float16 v16h;
typedef __attribute__((ext_vector_type(8)))  _Float16 v8h;
typedef __attribute__((ext_vector_type(8)))  float    v8f;

union V16 { v16h v; v8h h[2]; };

// CDNA5 async LDS staging: one b128 per lane, global -> LDS, no VGPR data path.
// Tracked with ASYNCcnt (s_wait_asynccnt), see CDNA5 ISA.
static __device__ __forceinline__ void async_copy_b128(unsigned lds_addr, const void* gaddr) {
  asm volatile("global_load_async_to_lds_b128 %0, %1, off"
               :: "v"(lds_addr), "v"(gaddr) : "memory");
}
static __device__ __forceinline__ void wait_async_all() {
  asm volatile("s_wait_asynccnt 0x0" ::: "memory");
}

// ---------------------------------------------------------------- fp32 -> f16
__global__ __launch_bounds__(256) void f32_to_f16_kernel(const float* __restrict__ in,
                                                         _Float16* __restrict__ out, int n) {
  int i = blockIdx.x * 256 + threadIdx.x;
  if (i < n) out[i] = (_Float16)in[i];
}

// ---------------------------------------------------------------- LayerNorm
__global__ __launch_bounds__(256) void ln_kernel(const float* __restrict__ s,
                                                 const float* __restrict__ lnw,
                                                 _Float16* __restrict__ nsh) {
  __shared__ float red[8];
  __shared__ float bc;
  const int row = blockIdx.x;
  const float* sr = s + (size_t)row * Csz;
  float v[6];
  float acc = 0.f;
#pragma unroll
  for (int k = 0; k < 6; ++k) { v[k] = sr[threadIdx.x + k * 256]; acc += v[k]; }
  for (int off = 16; off; off >>= 1) acc += __shfl_xor(acc, off, 32);
  if ((threadIdx.x & 31) == 0) red[threadIdx.x >> 5] = acc;
  __syncthreads();
  if (threadIdx.x == 0) { float t = 0.f; for (int i = 0; i < 8; ++i) t += red[i]; bc = t * (1.0f / Csz); }
  __syncthreads();
  const float mean = bc;
  float acc2 = 0.f;
#pragma unroll
  for (int k = 0; k < 6; ++k) { v[k] -= mean; acc2 += v[k] * v[k]; }
  for (int off = 16; off; off >>= 1) acc2 += __shfl_xor(acc2, off, 32);
  if ((threadIdx.x & 31) == 0) red[threadIdx.x >> 5] = acc2;
  __syncthreads();
  if (threadIdx.x == 0) { float t = 0.f; for (int i = 0; i < 8; ++i) t += red[i]; bc = rsqrtf(t * (1.0f / Csz) + LN_EPS); }
  __syncthreads();
  const float sc = bc;
#pragma unroll
  for (int k = 0; k < 6; ++k) {
    int c = threadIdx.x + k * 256;
    nsh[(size_t)row * Csz + c] = (_Float16)(v[k] * sc * lnw[c]);
  }
}

// ---------------------------------------------------------------- WMMA GEMM
// C[M,N] = A[M,K] * B[N,K]^T, all row-major, K contiguous for both operands.
// Block = 256 threads = 8 waves, each wave owns a 16x16 C tile (block: 32x64).
// Ping-pong double buffer (K % 64 == 0): two independent fragment register
// sets, no cross-buffer moves, 4 b128 loads always in flight behind each WMMA.
#define LOADA(dst, k) { (dst).h[0] = *(const v8h*)(arow + (k)); (dst).h[1] = *(const v8h*)(arow + (k) + 16); }
#define LOADB(dst, k) { (dst).h[0] = *(const v8h*)(brow + (k)); (dst).h[1] = *(const v8h*)(brow + (k) + 16); }

template <int K, int NT>
__global__ __launch_bounds__(256) void gemm_wmma_kernel(const _Float16* __restrict__ A,
                                                        const _Float16* __restrict__ Bm,
                                                        float* __restrict__ C) {
  static_assert((K & 63) == 0, "K must be a multiple of 64");
  const int wid  = threadIdx.x >> 5;
  const int lane = threadIdx.x & 31;
  const int hl   = lane >> 4;   // lane half: 0 or 1
  const int l    = lane & 15;
  const int m0 = blockIdx.y * 32 + (wid & 1) * 16;
  const int n0 = blockIdx.x * 64 + (wid >> 1) * 16;
  const _Float16* arow = A  + (size_t)(m0 + l) * K + hl * 8;   // A row M = m0+l
  const _Float16* brow = Bm + (size_t)(n0 + l) * K + hl * 8;   // B "col" N = n0+l
  v8f acc = {};
  V16 a0, b0, a1, b1;
  LOADA(a0, 0) LOADB(b0, 0)
  for (int k0 = 0; k0 < K; k0 += 64) {
    __builtin_prefetch(arow + k0 + 128, 0, 1);   // -> global_prefetch_b8
    __builtin_prefetch(brow + k0 + 128, 0, 1);
    LOADA(a1, k0 + 32) LOADB(b1, k0 + 32)
    acc = __builtin_amdgcn_wmma_f32_16x16x32_f16(false, a0.v, false, b0.v,
                                                 (short)0, acc, false, false);
    if (k0 + 64 < K) { LOADA(a0, k0 + 64) LOADB(b0, k0 + 64) }
    acc = __builtin_amdgcn_wmma_f32_16x16x32_f16(false, a1.v, false, b1.v,
                                                 (short)0, acc, false, false);
  }
  // C layout: VGPR r -> M = m0 + hl*8 + r, N = n0 + l
  float* crow = C + (size_t)(m0 + hl * 8) * NT + n0 + l;
#pragma unroll
  for (int r = 0; r < 8; ++r) crow[(size_t)r * NT] = acc[r];
}

// ---------------------------------------------------------------- geometry
__global__ __launch_bounds__(256) void geom_kernel(const float* __restrict__ p,
                                                   const float* __restrict__ rot,
                                                   const float* __restrict__ trans,
                                                   float* __restrict__ QR, float* __restrict__ KR,
                                                   float* __restrict__ VT, float* __restrict__ QD,
                                                   float* __restrict__ KD, float* __restrict__ Q2,
                                                   float* __restrict__ K2) {
  const int idx = blockIdx.x * 256 + threadIdx.x;
  if (idx >= Bsz * Ssz * 640) return;
  const int g  = idx % 640;
  const int bs = idx / 640;                // b*S + s
  const int b = bs / Ssz, sres = bs % Ssz;
  const float* R = rot + (size_t)bs * 9;
  const float* pr = p + (size_t)bs * DPROJ;
  if (g < 384) {
    const float* vv = pr + g * 3;
    float vx = vv[0], vy = vv[1], vz = vv[2];
    float r0 = R[0] * vx + R[1] * vy + R[2] * vz;
    float r1 = R[3] * vx + R[4] * vy + R[5] * vz;
    float r2 = R[6] * vx + R[7] * vy + R[8] * vz;
    const int head = g & 127;
    float* dst = (g < 128) ? QR : ((g < 256) ? KR : VT);
    size_t o = ((size_t)(b * Hsz + head) * Ssz + sres) * 3;
    dst[o] = r0; dst[o + 1] = r1; dst[o + 2] = r2;
  } else {
    const int gd = g - 384;
    const float* vv = pr + NROT + gd * 3;
    float vx = vv[0], vy = vv[1], vz = vv[2];
    const float* tr = trans + (size_t)bs * 3;
    float r0 = R[0] * vx + R[1] * vy + R[2] * vz + tr[0];
    float r1 = R[3] * vx + R[4] * vy + R[5] * vz + tr[1];
    float r2 = R[6] * vx + R[7] * vy + R[8] * vz + tr[2];
    const int head = gd & 127;
    float* dst  = (gd < 128) ? QD : KD;
    float* dst2 = (gd < 128) ? Q2 : K2;
    size_t o = (size_t)(b * Hsz + head) * Ssz + sres;
    dst[o * 3] = r0; dst[o * 3 + 1] = r1; dst[o * 3 + 2] = r2;
    dst2[o] = r0 * r0 + r1 * r1 + r2 * r2;
  }
}

// ---------------------------------------------------------------- fused attention
// One block per (b,h) and 8 query rows; K-side (contiguous per (b,h)) staged into LDS with
// CDNA5 GLOBAL_LOAD_ASYNC_TO_LDS_B128 (ASYNCcnt), bypassing VGPRs entirely.
// 8 waves each own one query row s; full softmax row in LDS; epilogue applies
// rot^T + affine mask and emits f16 ao for GEMM2.
__global__ __launch_bounds__(256) void attn_kernel(const float* __restrict__ QR, const float* __restrict__ KR,
                                                   const float* __restrict__ VT, const float* __restrict__ QD,
                                                   const float* __restrict__ KD, const float* __restrict__ Q2,
                                                   const float* __restrict__ K2, const float* __restrict__ rot,
                                                   const float* __restrict__ dist_scale,
                                                   const float* __restrict__ rot_scale,
                                                   const unsigned char* __restrict__ amask,
                                                   const int* __restrict__ seqid,
                                                   const int* __restrict__ chainid,
                                                   _Float16* __restrict__ aoh) {
  __shared__ __align__(16) float sKR[Ssz * 3];
  __shared__ __align__(16) float sKD[Ssz * 3];
  __shared__ __align__(16) float sVT[Ssz * 3];
  __shared__ __align__(16) float sK2[Ssz];
  __shared__ float sScore[8][Ssz];
  __shared__ int   sChain[Ssz];
  __shared__ int   sSeq[Ssz];
  __shared__ int   sMask[Ssz];

  const int bh = blockIdx.x;
  const int b = bh / Hsz, h = bh % Hsz;
  const size_t base = (size_t)bh * Ssz;

  // --- async bulk staging: 3 x 4608B + 1536B, 16B per lane per issue ---
  const float* gKR = KR + base * 3;
  const float* gKD = KD + base * 3;
  const float* gVT = VT + base * 3;
  const float* gK2 = K2 + base;
  for (int i = threadIdx.x; i < (Ssz * 3) / 4; i += 256) {   // 288 chunks
    async_copy_b128((unsigned)(uintptr_t)(&sKR[i * 4]), gKR + i * 4);
    async_copy_b128((unsigned)(uintptr_t)(&sKD[i * 4]), gKD + i * 4);
    async_copy_b128((unsigned)(uintptr_t)(&sVT[i * 4]), gVT + i * 4);
  }
  for (int i = threadIdx.x; i < Ssz / 4; i += 256) {         // 96 chunks
    async_copy_b128((unsigned)(uintptr_t)(&sK2[i * 4]), gK2 + i * 4);
  }
  // masks / ids (small, mixed dtypes) through the normal path
  for (int t = threadIdx.x; t < Ssz; t += 256) {
    sChain[t] = chainid[b * Ssz + t];
    sSeq[t]   = seqid[b * Ssz + t];
    sMask[t]  = (int)amask[b * Ssz + t];
  }
  wait_async_all();
  __syncthreads();

  const int w = threadIdx.x >> 5;
  const int lane = threadIdx.x & 31;
  const int srow = blockIdx.y * 8 + w;
  const size_t qo = (base + srow) * 3;
  const float q0 = QR[qo], q1 = QR[qo + 1], q2v = QR[qo + 2];
  const float d0 = QD[qo], d1 = QD[qo + 1], d2 = QD[qo + 2];
  const float qq = Q2[base + srow];
  const int cs = chainid[b * Ssz + srow];
  const int ss = seqid[b * Ssz + srow];
  const float rw = log1pf(__expf(rot_scale[h]));
  const float dw = log1pf(__expf(dist_scale[h]));

  float lmax = NEGF;
#pragma unroll
  for (int it = 0; it < Ssz / 32; ++it) {
    const int t = it * 32 + lane;
    const float rdot = q0 * sKR[t * 3] + q1 * sKR[t * 3 + 1] + q2v * sKR[t * 3 + 2];
    const float kdd  = d0 * sKD[t * 3] + d1 * sKD[t * 3 + 1] + d2 * sKD[t * 3 + 2];
    const float dsq  = qq + sK2[t] - 2.0f * kdd;
    const float dist = sqrtf(fmaxf(dsq, 0.0f)) * INV_SQRT3;
    float bias;
    if (sChain[t] != cs || !sMask[t]) bias = NEGF;
    else bias = (sSeq[t] == ss) ? 1.0f : 0.0f;
    const float sc = rdot * INV_SQRT3 * rw - dist * dw + bias;
    sScore[w][t] = sc;
    lmax = fmaxf(lmax, sc);
  }
  for (int off = 16; off; off >>= 1) lmax = fmaxf(lmax, __shfl_xor(lmax, off, 32));

  float sum = 0.f, o0 = 0.f, o1 = 0.f, o2 = 0.f;
#pragma unroll
  for (int it = 0; it < Ssz / 32; ++it) {
    const int t = it * 32 + lane;
    const float e = __expf(sScore[w][t] - lmax);
    sum += e;
    o0 += e * sVT[t * 3];
    o1 += e * sVT[t * 3 + 1];
    o2 += e * sVT[t * 3 + 2];
  }
  for (int off = 16; off; off >>= 1) {
    sum += __shfl_xor(sum, off, 32);
    o0  += __shfl_xor(o0, off, 32);
    o1  += __shfl_xor(o1, off, 32);
    o2  += __shfl_xor(o2, off, 32);
  }
  if (lane == 0) {
    const float inv = 1.0f / sum;
    const float v0 = o0 * inv, v1 = o1 * inv, v2 = o2 * inv;
    const float* R = rot + (size_t)(b * Ssz + srow) * 9;   // apply R^T
    const float m = amask[b * Ssz + srow] ? 1.0f : 0.0f;
    const float a0 = (R[0] * v0 + R[3] * v1 + R[6] * v2) * m;
    const float a1 = (R[1] * v0 + R[4] * v1 + R[7] * v2) * m;
    const float a2 = (R[2] * v0 + R[5] * v1 + R[8] * v2) * m;
    _Float16* dst = aoh + (size_t)(b * Ssz + srow) * AODIM + h * 3;
    dst[0] = (_Float16)a0; dst[1] = (_Float16)a1; dst[2] = (_Float16)a2;
  }
}

// ---------------------------------------------------------------- launch
extern "C" void kernel_launch(void* const* d_in, const int* in_sizes, int n_in,
                              void* d_out, int out_size, void* d_ws, size_t ws_size,
                              hipStream_t stream) {
  (void)in_sizes; (void)n_in; (void)out_size; (void)ws_size;
  const float* s_in   = (const float*)d_in[0];
  const float* rot    = (const float*)d_in[1];
  const float* trans  = (const float*)d_in[2];
  const float* lnw    = (const float*)d_in[3];
  const float* projw  = (const float*)d_in[4];
  const float* outw   = (const float*)d_in[5];
  const float* dsc    = (const float*)d_in[6];
  const float* rsc    = (const float*)d_in[7];
  const unsigned char* amask = (const unsigned char*)d_in[8];
  const int* seqid    = (const int*)d_in[9];
  const int* chainid  = (const int*)d_in[10];
  float* out = (float*)d_out;

  char* w = (char*)d_ws;
  size_t off = 0;
  auto alloc = [&](size_t bytes) -> void* {
    void* ptr = w + off;
    off += (bytes + 255) & ~(size_t)255;
    return ptr;
  };
  _Float16* nsh = (_Float16*)alloc((size_t)ROWS * Csz * 2);
  _Float16* pwh = (_Float16*)alloc((size_t)DPROJ * Csz * 2);
  _Float16* owh = (_Float16*)alloc((size_t)Csz * AODIM * 2);
  float* p  = (float*)alloc((size_t)ROWS * DPROJ * 4);
  float* QR = (float*)alloc((size_t)Bsz * Hsz * Ssz * 3 * 4);
  float* KR = (float*)alloc((size_t)Bsz * Hsz * Ssz * 3 * 4);
  float* VT = (float*)alloc((size_t)Bsz * Hsz * Ssz * 3 * 4);
  float* QD = (float*)alloc((size_t)Bsz * Hsz * Ssz * 3 * 4);
  float* KD = (float*)alloc((size_t)Bsz * Hsz * Ssz * 3 * 4);
  float* Q2 = (float*)alloc((size_t)Bsz * Hsz * Ssz * 4);
  float* K2 = (float*)alloc((size_t)Bsz * Hsz * Ssz * 4);
  _Float16* aoh = (_Float16*)alloc((size_t)ROWS * AODIM * 2);

  // 1. weight converts (everything is L2-resident on a 192MB-L2 part)
  f32_to_f16_kernel<<<(DPROJ * Csz + 255) / 256, 256, 0, stream>>>(projw, pwh, DPROJ * Csz);
  f32_to_f16_kernel<<<(Csz * AODIM + 255) / 256, 256, 0, stream>>>(outw, owh, Csz * AODIM);
  // 2. LayerNorm -> f16 activations
  ln_kernel<<<ROWS, 256, 0, stream>>>(s_in, lnw, nsh);
  // 3. GEMM1: p = ns @ proj_w^T  (768 x 1920, K=1536) via v_wmma_f32_16x16x32_f16
  gemm_wmma_kernel<Csz, DPROJ><<<dim3(DPROJ / 64, ROWS / 32), 256, 0, stream>>>(nsh, pwh, p);
  // 4. rotations / points / squared norms
  geom_kernel<<<(Bsz * Ssz * 640 + 255) / 256, 256, 0, stream>>>(p, rot, trans,
                                                                 QR, KR, VT, QD, KD, Q2, K2);
  // 5. fused attention: async-LDS staged K-side, scores never hit HBM
  attn_kernel<<<dim3(Bsz * Hsz, Ssz / 8), 256, 0, stream>>>(QR, KR, VT, QD, KD, Q2, K2,
                                                            rot, dsc, rsc, amask, seqid,
                                                            chainid, aoh);
  // 6. GEMM2: out = ao @ out_w^T (768 x 1536, K=384) via WMMA
  gemm_wmma_kernel<AODIM, Csz><<<dim3(Csz / 64, ROWS / 32), 256, 0, stream>>>(aoh, owh, out);
}
